// QuantumAlphaZeroNet_63273458205141
// MI455X (gfx1250) — compile-verified, matrix operand order checked by reading, not run on hardware
//
#include <hip/hip_runtime.h>
#include <hip/hip_bf16.h>

typedef __attribute__((ext_vector_type(16))) _Float16 v16h;
typedef __attribute__((ext_vector_type(8)))  _Float16 v8h;
typedef __attribute__((ext_vector_type(8)))  float    v8f;

#define EPSV 1e-5f

// ===========================================================================
// WMMA fragment layouts (CDNA5 ISA 7.12.2, 16-bit, wave32):
//   A (16x32): M = lane&15, K = (j&7) + (j>>3)*16 + (lane>>4)*8
//   B (32x16): N = lane&15, K = (j&7) + (j>>3)*16 + (lane>>4)*8
//   D (16x16): N = lane&15, M = r + 8*(lane>>4)
// Packed operand buffers hold, per fragment, 32 lanes x 16 halves (1 KB):
//   PA frag = coTile*nK + kc ; PB frag = (img*nK + kc)*23 + tile
// For a thread pair (lane, jg in {0,1}), the 8 halves j = jg*8..jg*8+7 have
// CONSECUTIVE K: k = kc*32 + jg*16 + (lane>>4)*8 + jj  (both A and B layouts).
// ===========================================================================

// ---------------------------------------------------------------------------
// Weight pack: OIHW f32 -> A-fragment f16.  One thread = 8 halves (b128 store).
// ---------------------------------------------------------------------------
__global__ void wpack(const float* __restrict__ W, _Float16* __restrict__ PA,
                      int Co, int Kdim, int nK, int nCoT) {
  int t = blockIdx.x * blockDim.x + threadIdx.x;
  int total = nCoT * nK * 64;
  if (t >= total) return;
  int u = t & 63, frag = t >> 6;
  int lane = u >> 1, jg = u & 1;
  int kc = frag % nK, coT = frag / nK;
  int co = (coT << 4) + (lane & 15);
  int k0 = (kc << 5) + (jg << 4) + ((lane >> 4) << 3);
  int coc = co < Co ? co : Co - 1;
  v8h vals;
#pragma unroll
  for (int jj = 0; jj < 8; ++jj) {
    int k = k0 + jj;
    int kcl = k < Kdim ? k : Kdim - 1;
    float v = W[(size_t)coc * Kdim + kcl];
    vals[jj] = (co < Co && k < Kdim) ? (_Float16)v : (_Float16)0.f;
  }
  *(v8h*)(PA + ((size_t)frag * 32 + lane) * 16 + (jg << 3)) = vals;
}

// ---------------------------------------------------------------------------
// im2col pack: f32 NCHW -> B-fragment f16. Branch-free: clamped addresses +
// select-zero (no exec-mask churn). One thread = 8 halves (one b128 store).
// ---------------------------------------------------------------------------
__global__ void im2col_pack(const float* __restrict__ X, _Float16* __restrict__ PB,
                            int C, int nK, int nImgs) {
  int t = blockIdx.x * blockDim.x + threadIdx.x;
  int total = nImgs * nK * 23 * 64;
  if (t >= total) return;
  int u = t & 63, frag = t >> 6;
  int lane = u >> 1, jg = u & 1;
  int tile = frag % 23;
  int tmp  = frag / 23;
  int kc = tmp % nK, img = tmp / nK;
  int p = (tile << 4) + (lane & 15);
  bool pOK = p < 361;
  int pc = pOK ? p : 360;
  int y = pc / 19, x = pc - y * 19;
  int k0 = (kc << 5) + (jg << 4) + ((lane >> 4) << 3);
  const float* xb = X + (size_t)img * C * 361;
  v8h vals;
#pragma unroll
  for (int jj = 0; jj < 8; ++jj) {
    int k = k0 + jj;
    int ci = k / 9;
    int r  = k - ci * 9;
    int ky = r / 3, kx = r - ky * 3;
    int iy = y + ky - 1, ix = x + kx - 1;
    bool ok = pOK && ci < C && (unsigned)iy < 19u && (unsigned)ix < 19u;
    int cic = ci < C ? ci : 0;
    int iyc = (unsigned)iy < 19u ? iy : 0;
    int ixc = (unsigned)ix < 19u ? ix : 0;
    float v = xb[(size_t)cic * 361 + iyc * 19 + ixc];
    vals[jj] = ok ? (_Float16)v : (_Float16)0.f;
  }
  *(v8h*)(PB + ((size_t)frag * 32 + lane) * 16 + (jg << 3)) = vals;
}

// ---------------------------------------------------------------------------
// im2col pack with fused LayerNorm+ReLU (per-element gamma/beta (C,361)).
// ---------------------------------------------------------------------------
__global__ void ln_im2col_pack(const float* __restrict__ X, const float* __restrict__ g,
                               const float* __restrict__ bb, const float* __restrict__ mu,
                               const float* __restrict__ rs, _Float16* __restrict__ PB,
                               int C, int nK, int nImgs) {
  int t = blockIdx.x * blockDim.x + threadIdx.x;
  int total = nImgs * nK * 23 * 64;
  if (t >= total) return;
  int u = t & 63, frag = t >> 6;
  int lane = u >> 1, jg = u & 1;
  int tile = frag % 23;
  int tmp  = frag / 23;
  int kc = tmp % nK, img = tmp / nK;
  int p = (tile << 4) + (lane & 15);
  bool pOK = p < 361;
  int pc = pOK ? p : 360;
  int y = pc / 19, x = pc - y * 19;
  int k0 = (kc << 5) + (jg << 4) + ((lane >> 4) << 3);
  const float* xb = X + (size_t)img * C * 361;
  float m = mu[img], rr = rs[img];
  v8h vals;
#pragma unroll
  for (int jj = 0; jj < 8; ++jj) {
    int k = k0 + jj;
    int ci = k / 9;
    int r  = k - ci * 9;
    int ky = r / 3, kx = r - ky * 3;
    int iy = y + ky - 1, ix = x + kx - 1;
    bool ok = pOK && ci < C && (unsigned)iy < 19u && (unsigned)ix < 19u;
    int cic = ci < C ? ci : 0;
    int iyc = (unsigned)iy < 19u ? iy : 0;
    int ixc = (unsigned)ix < 19u ? ix : 0;
    int cp = cic * 361 + iyc * 19 + ixc;
    float v = (xb[cp] - m) * rr * g[cp] + bb[cp];
    v = fmaxf(v, 0.f);
    vals[jj] = ok ? (_Float16)v : (_Float16)0.f;
  }
  *(v8h*)(PB + ((size_t)frag * 32 + lane) * 16 + (jg << 3)) = vals;
}

// ---------------------------------------------------------------------------
// Conv 3x3 as pure fragment-GEMM: 2x 32B loads + 1 WMMA per K-chunk.
// Grid: (imgs, Co/16, ceil(23/4)); block (32,4) = 4 waves (wave32).
// ---------------------------------------------------------------------------
__global__ void conv3x3_wmma(const _Float16* __restrict__ PB, const _Float16* __restrict__ PA,
                             float* __restrict__ Y, int Co, int nK) {
  const int img  = blockIdx.x;
  const int coT  = blockIdx.y;
  const int tile = (blockIdx.z << 2) + threadIdx.y;
  if (tile >= 23) return;                 // uniform per wave
  const int lane = threadIdx.x;
  const v16h* ap = (const v16h*)PA + ((size_t)coT * nK) * 32 + lane;
  const v16h* bp = (const v16h*)PB + (((size_t)img * nK) * 23 + tile) * 32 + lane;
  v8f acc = {};
#pragma unroll 2
  for (int kc = 0; kc < nK; ++kc) {
    v16h a = ap[(size_t)kc * 32];
    v16h b = bp[(size_t)kc * 736];        // 23 tiles * 32 lanes
    acc = __builtin_amdgcn_wmma_f32_16x16x32_f16(false, a, false, b, (short)0, acc,
                                                 false, false);
  }
  const int p = (tile << 4) + (lane & 15);
  if (p < 361) {
    const size_t obase = ((size_t)img * Co) * 361 + p;
    const int mHi = (lane >> 4) << 3;
#pragma unroll
    for (int r = 0; r < 8; ++r) {
      int c2 = (coT << 4) + mHi + r;
      if (c2 < Co) Y[obase + (size_t)c2 * 361] = acc[r];
    }
  }
}

// ---------------------------------------------------------------------------
// 1x1 conv (scalar; tiny FLOPs)
// ---------------------------------------------------------------------------
__global__ void conv1x1_k(const float* __restrict__ X, const float* __restrict__ W,
                          float* __restrict__ Y, int Ci, int Co, int P, int total) {
  int i = blockIdx.x * blockDim.x + threadIdx.x;
  if (i >= total) return;
  int img = i / (Co * P);
  int rem = i - img * (Co * P);
  int co  = rem / P;
  int p   = rem - co * P;
  const float* xp = X + ((size_t)img * Ci) * P + p;
  const float* wp = W + (size_t)co * Ci;
  float s = 0.f;
#pragma unroll 8
  for (int ci = 0; ci < Ci; ++ci) s += xp[(size_t)ci * P] * wp[ci];
  Y[i] = s;
}

// ---------------------------------------------------------------------------
// LayerNorm stats per image (over C*P). grid.x = #imgs.
// ---------------------------------------------------------------------------
__global__ void ln_stats(const float* __restrict__ x, float* __restrict__ mu,
                         float* __restrict__ rs, int C, int P) {
  __shared__ float ss[256], sq[256];
  const int img = blockIdx.x;
  const int n = C * P;
  const float* xp = x + (size_t)img * n;
  float s = 0.f, q = 0.f;
  for (int i = threadIdx.x; i < n; i += 256) {
    float v = xp[i];
    s += v; q += v * v;
  }
  ss[threadIdx.x] = s; sq[threadIdx.x] = q;
  __syncthreads();
  for (int off = 128; off > 0; off >>= 1) {
    if (threadIdx.x < off) { ss[threadIdx.x] += ss[threadIdx.x + off];
                             sq[threadIdx.x] += sq[threadIdx.x + off]; }
    __syncthreads();
  }
  if (threadIdx.x == 0) {
    float m = ss[0] / n;
    float v = sq[0] / n - m * m;
    mu[img] = m;
    rs[img] = rsqrtf(v + EPSV);
  }
}

// ---------------------------------------------------------------------------
// BatchNorm stats per channel (over N*P). grid.x = C.
// ---------------------------------------------------------------------------
__global__ void bn_stats(const float* __restrict__ x, float* __restrict__ mu,
                         float* __restrict__ rs, int N, int C, int P) {
  __shared__ float ss[256], sq[256];
  const int c = blockIdx.x;
  const int n = N * P;
  float s = 0.f, q = 0.f;
  for (int i = threadIdx.x; i < n; i += 256) {
    int nn = i / P, p = i - nn * P;
    float v = x[((size_t)nn * C + c) * P + p];
    s += v; q += v * v;
  }
  ss[threadIdx.x] = s; sq[threadIdx.x] = q;
  __syncthreads();
  for (int off = 128; off > 0; off >>= 1) {
    if (threadIdx.x < off) { ss[threadIdx.x] += ss[threadIdx.x + off];
                             sq[threadIdx.x] += sq[threadIdx.x + off]; }
    __syncthreads();
  }
  if (threadIdx.x == 0) {
    float m = ss[0] / n;
    float v = sq[0] / n - m * m;
    mu[c] = m;
    rs[c] = rsqrtf(v + EPSV);
  }
}

// ---------------------------------------------------------------------------
// Candidate build: relu(LN(t2)) + residual -> cand_f (f32 NCHW).
// t2: (192 imgs = b*3+k, 384 ch = bw*128+cc). cand img = b*9 + k*3 + bw.
// ---------------------------------------------------------------------------
__global__ void cand_build(const float* __restrict__ t2, const float* __restrict__ g,
                           const float* __restrict__ bb, const float* __restrict__ mu,
                           const float* __restrict__ rs, const float* __restrict__ beam_f,
                           float* __restrict__ cand_f, int total) {
  int i = blockIdx.x * blockDim.x + threadIdx.x;
  if (i >= total) return;
  const int cpn = 384 * 361;
  int img = i / cpn;                 // b*3 + k
  int cp  = i - img * cpn;
  int c   = cp / 361;
  int p   = cp - c * 361;
  float v = (t2[i] - mu[img]) * rs[img] * g[cp] + bb[cp];
  v = fmaxf(v, 0.f);
  int bw = c >> 7;
  int cc = c & 127;
  int b_ = img / 3;
  int k  = img - b_ * 3;
  float z = beam_f[((size_t)img * 128 + cc) * 361 + p];
  size_t o = (((size_t)(b_ * 9 + k * 3 + bw)) * 128 + cc) * 361 + p;
  cand_f[o] = v + z;
}

// ---------------------------------------------------------------------------
// Fitness score: mean over (3,361) of relu(LN(f2) + zf3). grid.x = 576 imgs.
// ---------------------------------------------------------------------------
__global__ void fitness_score(const float* __restrict__ f2, const float* __restrict__ zf3,
                              const float* __restrict__ g, const float* __restrict__ bb,
                              const float* __restrict__ mu, const float* __restrict__ rs,
                              float* __restrict__ scores) {
  __shared__ float ss[128];
  const int img = blockIdx.x;
  const int n = 3 * 361;
  const float m = mu[img], r = rs[img];
  const float* fp = f2 + (size_t)img * n;
  const float* zp = zf3 + (size_t)img * n;
  float s = 0.f;
  for (int i = threadIdx.x; i < n; i += 128) {
    float v = (fp[i] - m) * r * g[i] + bb[i] + zp[i];
    s += fmaxf(v, 0.f);
  }
  ss[threadIdx.x] = s;
  __syncthreads();
  for (int off = 64; off > 0; off >>= 1) {
    if (threadIdx.x < off) ss[threadIdx.x] += ss[threadIdx.x + off];
    __syncthreads();
  }
  if (threadIdx.x == 0) scores[img] = ss[0] / n;
}

// ---------------------------------------------------------------------------
// Top-3 of 9 per batch (ordered, first occurrence on ties like lax.top_k).
// ---------------------------------------------------------------------------
__global__ void topk3(const float* __restrict__ scores, int* __restrict__ idx) {
  int b = threadIdx.x;
  if (b >= 64) return;
  float s[9];
  bool used[9];
#pragma unroll
  for (int i = 0; i < 9; ++i) { s[i] = scores[b * 9 + i]; used[i] = false; }
#pragma unroll
  for (int k = 0; k < 3; ++k) {
    float best = -3.4e38f; int bi = 0;
#pragma unroll
    for (int i = 0; i < 9; ++i)
      if (!used[i] && s[i] > best) { best = s[i]; bi = i; }
    used[bi] = true;
    idx[b * 3 + k] = bi;
  }
}

// ---------------------------------------------------------------------------
// Gather beam = cand[b, idx[b,k]] (f32).
// ---------------------------------------------------------------------------
__global__ void gather_beam(const float* __restrict__ cand_f, const int* __restrict__ idx,
                            float* __restrict__ beam_f, int total) {
  int i = blockIdx.x * blockDim.x + threadIdx.x;
  if (i >= total) return;
  const int cpn = 128 * 361;
  int bk = i / cpn;
  int cp = i - bk * cpn;
  int b_ = bk / 3;
  int k  = bk - b_ * 3;
  int sel = idx[b_ * 3 + k];
  beam_f[i] = cand_f[((size_t)(b_ * 9 + sel)) * cpn + cp];
}

// ---------------------------------------------------------------------------
// Stem: BN -> ReLU, replicated 3x into beam_f.
// ---------------------------------------------------------------------------
__global__ void stem_apply_rep(const float* __restrict__ x, const float* __restrict__ g,
                               const float* __restrict__ bb, const float* __restrict__ mu,
                               const float* __restrict__ rs, float* __restrict__ beam_f,
                               int total) {
  int i = blockIdx.x * blockDim.x + threadIdx.x;
  if (i >= total) return;
  const int cpn = 128 * 361;
  int b_ = i / cpn;
  int cp = i - b_ * cpn;
  int c  = cp / 361;
  float v = (x[i] - mu[c]) * rs[c] * g[c] + bb[c];
  v = fmaxf(v, 0.f);
#pragma unroll
  for (int k = 0; k < 3; ++k)
    beam_f[((size_t)(b_ * 3 + k)) * cpn + cp] = v;
}

// ---------------------------------------------------------------------------
// BN apply (+ReLU), per-channel scalar gamma/beta. In-place safe.
// ---------------------------------------------------------------------------
__global__ void bn_apply_relu(const float* __restrict__ x, const float* __restrict__ g,
                              const float* __restrict__ bb, const float* __restrict__ mu,
                              const float* __restrict__ rs, float* __restrict__ y,
                              int C, int P, int total) {
  int i = blockIdx.x * blockDim.x + threadIdx.x;
  if (i >= total) return;
  int cpn = C * P;
  int cp = i - (i / cpn) * cpn;
  int c = cp / P;
  float v = (x[i] - mu[c]) * rs[c] * g[c] + bb[c];
  y[i] = fmaxf(v, 0.f);
}

__global__ void extract_best(const float* __restrict__ beam_f, float* __restrict__ best,
                             int total) {
  int i = blockIdx.x * blockDim.x + threadIdx.x;
  if (i >= total) return;
  const int cpn = 128 * 361;
  int b_ = i / cpn;
  int cp = i - b_ * cpn;
  best[i] = beam_f[((size_t)(b_ * 3)) * cpn + cp];
}

// ---------------------------------------------------------------------------
// Dense FC: Y[b,o] = act(bias[o] + X[b,:] . W[o,:])
// ---------------------------------------------------------------------------
__global__ void fc_kernel(const float* __restrict__ X, const float* __restrict__ W,
                          const float* __restrict__ bias, float* __restrict__ Y,
                          int in_dim, int out_dim, int do_relu, int total) {
  int i = blockIdx.x * blockDim.x + threadIdx.x;
  if (i >= total) return;
  int b = i / out_dim;
  int o = i - b * out_dim;
  const float* xp = X + (size_t)b * in_dim;
  const float* wp = W + (size_t)o * in_dim;
  float s = bias[o];
#pragma unroll 8
  for (int j = 0; j < in_dim; ++j) s += xp[j] * wp[j];
  Y[i] = do_relu ? fmaxf(s, 0.f) : s;
}

__global__ void fc_tanh(const float* __restrict__ X, const float* __restrict__ W,
                        const float* __restrict__ bias, float* __restrict__ Y, int in_dim) {
  int b = threadIdx.x;
  if (b >= 64) return;
  const float* xp = X + (size_t)b * in_dim;
  float s = bias[0];
#pragma unroll 8
  for (int j = 0; j < in_dim; ++j) s += xp[j] * W[j];
  Y[b] = tanhf(s);
}

// ---------------------------------------------------------------------------
// Host launcher
// ---------------------------------------------------------------------------
static inline int blks(long n) { return (int)((n + 255) / 256); }

extern "C" void kernel_launch(void* const* d_in, const int* in_sizes, int n_in,
                              void* d_out, int out_size, void* d_ws, size_t ws_size,
                              hipStream_t stream) {
  (void)in_sizes; (void)n_in; (void)out_size; (void)ws_size;
  const float* x     = (const float*)d_in[0];
  const float* w_stem= (const float*)d_in[1];
  const float* bn_g  = (const float*)d_in[2];
  const float* bn_b  = (const float*)d_in[3];
  const float* tw1   = (const float*)d_in[4];
  const float* tln1g = (const float*)d_in[5];
  const float* tln1b = (const float*)d_in[6];
  const float* tw2   = (const float*)d_in[7];
  const float* tln2g = (const float*)d_in[8];
  const float* tln2b = (const float*)d_in[9];
  const float* fw1   = (const float*)d_in[10];
  const float* fln1g = (const float*)d_in[11];
  const float* fln1b = (const float*)d_in[12];
  const float* fw2   = (const float*)d_in[13];
  const float* fln2g = (const float*)d_in[14];
  const float* fln2b = (const float*)d_in[15];
  const float* fw3   = (const float*)d_in[16];
  const float* pw    = (const float*)d_in[17];
  const float* pbg   = (const float*)d_in[18];
  const float* pbb   = (const float*)d_in[19];
  const float* plw   = (const float*)d_in[20];
  const float* plb   = (const float*)d_in[21];
  const float* vw    = (const float*)d_in[22];
  const float* vbg   = (const float*)d_in[23];
  const float* vbb   = (const float*)d_in[24];
  const float* vl1w  = (const float*)d_in[25];
  const float* vl1b  = (const float*)d_in[26];
  const float* vl2w  = (const float*)d_in[27];
  const float* vl2b  = (const float*)d_in[28];

  // ---- workspace layout ----
  char* base = (char*)d_ws;
  size_t off = 0;
  auto alloc = [&](size_t bytes) -> char* {
    char* p = base + off;
    off = (off + bytes + 255) & ~(size_t)255;
    return p;
  };
  const long N_STEM = 64L * 128 * 361;         // 2,957,312
  const long N_BEAM = 192L * 128 * 361;        // 8,871,936
  const long N_T2   = 192L * 384 * 361;        // 26,615,808 (== 576*128*361)
  const long N_F2   = 576L * 3 * 361;          // 623,808
  const long PB_FRAGS = 144L * 36 * 23;        // max chunk: 144 imgs, nK=36

  _Float16* PAstem = (_Float16*)alloc(8L * 6 * 512 * 2);
  _Float16* PAtw1  = (_Float16*)alloc(8L * 36 * 512 * 2);
  _Float16* PAtw2  = (_Float16*)alloc(24L * 36 * 512 * 2);
  _Float16* PAfw1  = (_Float16*)alloc(8L * 36 * 512 * 2);
  _Float16* PAfw2  = (_Float16*)alloc(1L * 36 * 512 * 2);
  _Float16* PB     = (_Float16*)alloc(PB_FRAGS * 512 * 2);   // ~116 MB, reused
  float*    stemO  = (float*)alloc(N_STEM * 4);
  float*    ch_mu  = (float*)alloc(128 * 4);
  float*    ch_rs  = (float*)alloc(128 * 4);
  float*    img_mu = (float*)alloc(576 * 4);
  float*    img_rs = (float*)alloc(576 * 4);
  float*    beam_f = (float*)alloc(N_BEAM * 4);
  float*    t1     = (float*)alloc(N_BEAM * 4);
  float*    t2     = (float*)alloc(N_T2 * 4);
  float*    f1     = t2;                        // alias: t2 dead once cand built
  float*    cand_f = (float*)alloc(N_T2 * 4);
  float*    f2     = (float*)alloc(N_F2 * 4);
  float*    zf3    = (float*)alloc(N_F2 * 4);
  float*    scores = (float*)alloc(576 * 4);
  int*      idxb   = (int*)alloc(192 * 4);
  float*    best   = (float*)alloc(N_STEM * 4);
  float*    ph     = (float*)alloc(64L * 2 * 361 * 4);
  float*    vh     = (float*)alloc(64L * 361 * 4);
  float*    v1     = (float*)alloc(64L * 256 * 4);

  float* pi_out  = (float*)d_out;               // (64,362)
  float* val_out = (float*)d_out + 64 * 362;    // (64,)

  const dim3 WB(32, 4, 1);

  // ---- pack weights once (constant across depth loop) ----
  wpack<<<blks(8L * 6 * 64), 256, 0, stream>>>(w_stem, PAstem, 128, 162, 6, 8);
  wpack<<<blks(8L * 36 * 64), 256, 0, stream>>>(tw1, PAtw1, 128, 1152, 36, 8);
  wpack<<<blks(24L * 36 * 64), 256, 0, stream>>>(tw2, PAtw2, 384, 1152, 36, 24);
  wpack<<<blks(8L * 36 * 64), 256, 0, stream>>>(fw1, PAfw1, 128, 1152, 36, 8);
  wpack<<<blks(1L * 36 * 64), 256, 0, stream>>>(fw2, PAfw2, 3, 1152, 36, 1);

  // ---- stem: pack -> conv -> BN -> ReLU -> beam(x3) ----
  im2col_pack<<<blks(64L * 6 * 23 * 64), 256, 0, stream>>>(x, PB, 18, 6, 64);
  conv3x3_wmma<<<dim3(64, 8, 6), WB, 0, stream>>>(PB, PAstem, stemO, 128, 6);
  bn_stats<<<128, 256, 0, stream>>>(stemO, ch_mu, ch_rs, 64, 128, 361);
  stem_apply_rep<<<blks(N_STEM), 256, 0, stream>>>(stemO, bn_g, bn_b, ch_mu, ch_rs,
                                                   beam_f, (int)N_STEM);

  // ---- beam search ----
  const long PACK96 = 96L * 36 * 23 * 64;
  const long PACK144 = 144L * 36 * 23 * 64;
  for (int d = 0; d < 10; ++d) {
    // transition conv1: beam_f (192 imgs, 2 chunks of 96)
    for (int c = 0; c < 2; ++c) {
      long io = (long)c * 96 * 128 * 361;
      im2col_pack<<<blks(PACK96), 256, 0, stream>>>(beam_f + io, PB, 128, 36, 96);
      conv3x3_wmma<<<dim3(96, 8, 6), WB, 0, stream>>>(PB, PAtw1, t1 + io, 128, 36);
    }
    ln_stats<<<192, 256, 0, stream>>>(t1, img_mu, img_rs, 128, 361);
    // transition conv2: LN(t1)+ReLU fused into pack (2 chunks of 96)
    for (int c = 0; c < 2; ++c) {
      long io = (long)c * 96 * 128 * 361;
      long oo = (long)c * 96 * 384 * 361;
      ln_im2col_pack<<<blks(PACK96), 256, 0, stream>>>(t1 + io, tln1g, tln1b,
                                                       img_mu + c * 96, img_rs + c * 96,
                                                       PB, 128, 36, 96);
      conv3x3_wmma<<<dim3(96, 24, 6), WB, 0, stream>>>(PB, PAtw2, t2 + oo, 384, 36);
    }
    ln_stats<<<192, 256, 0, stream>>>(t2, img_mu, img_rs, 384, 361);
    cand_build<<<blks(N_T2), 256, 0, stream>>>(t2, tln2g, tln2b, img_mu, img_rs,
                                               beam_f, cand_f, (int)N_T2);
    // fitness conv1: cand_f (576 imgs, 4 chunks of 144)
    for (int c = 0; c < 4; ++c) {
      long io = (long)c * 144 * 128 * 361;
      im2col_pack<<<blks(PACK144), 256, 0, stream>>>(cand_f + io, PB, 128, 36, 144);
      conv3x3_wmma<<<dim3(144, 8, 6), WB, 0, stream>>>(PB, PAfw1, f1 + io, 128, 36);
    }
    ln_stats<<<576, 256, 0, stream>>>(f1, img_mu, img_rs, 128, 361);
    // fitness conv2: LN(f1)+ReLU fused into pack (4 chunks of 144)
    for (int c = 0; c < 4; ++c) {
      long io = (long)c * 144 * 128 * 361;
      long oo = (long)c * 144 * 3 * 361;
      ln_im2col_pack<<<blks(PACK144), 256, 0, stream>>>(f1 + io, fln1g, fln1b,
                                                        img_mu + c * 144, img_rs + c * 144,
                                                        PB, 128, 36, 144);
      conv3x3_wmma<<<dim3(144, 1, 6), WB, 0, stream>>>(PB, PAfw2, f2 + oo, 3, 36);
    }
    ln_stats<<<576, 256, 0, stream>>>(f2, img_mu, img_rs, 3, 361);
    conv1x1_k<<<blks(N_F2), 256, 0, stream>>>(cand_f, fw3, zf3, 128, 3, 361, (int)N_F2);
    fitness_score<<<576, 128, 0, stream>>>(f2, zf3, fln2g, fln2b, img_mu, img_rs, scores);
    // select
    topk3<<<1, 64, 0, stream>>>(scores, idxb);
    gather_beam<<<blks(N_BEAM), 256, 0, stream>>>(cand_f, idxb, beam_f, (int)N_BEAM);
  }

  // ---- heads ----
  extract_best<<<blks(N_STEM), 256, 0, stream>>>(beam_f, best, (int)N_STEM);

  // policy
  conv1x1_k<<<blks(64L * 2 * 361), 256, 0, stream>>>(best, pw, ph, 128, 2, 361, 64 * 2 * 361);
  bn_stats<<<2, 256, 0, stream>>>(ph, ch_mu, ch_rs, 64, 2, 361);
  bn_apply_relu<<<blks(64L * 2 * 361), 256, 0, stream>>>(ph, pbg, pbb, ch_mu, ch_rs,
                                                         ph, 2, 361, 64 * 2 * 361);
  fc_kernel<<<blks(64L * 362), 256, 0, stream>>>(ph, plw, plb, pi_out, 722, 362, 0, 64 * 362);

  // value
  conv1x1_k<<<blks(64L * 361), 256, 0, stream>>>(best, vw, vh, 128, 1, 361, 64 * 361);
  bn_stats<<<1, 256, 0, stream>>>(vh, ch_mu, ch_rs, 64, 1, 361);
  bn_apply_relu<<<blks(64L * 361), 256, 0, stream>>>(vh, vbg, vbb, ch_mu, ch_rs,
                                                     vh, 1, 361, 64 * 361);
  fc_kernel<<<blks(64L * 256), 256, 0, stream>>>(vh, vl1w, vl1b, v1, 361, 256, 1, 64 * 256);
  fc_tanh<<<1, 64, 0, stream>>>(v1, vl2w, vl2b, val_out, 256);
}
